// ResidualMamba_21560735825985
// MI455X (gfx1250) — matrix operand order repaired and checked by reading, not compiled, not run on hardware
//
#include <hip/hip_runtime.h>
#include <hip/hip_bf16.h>
#include <math.h>

typedef __attribute__((ext_vector_type(8)))  _Float16 v8h;
typedef __attribute__((ext_vector_type(16))) _Float16 v16h;
typedef __attribute__((ext_vector_type(8)))  float    v8f;

constexpr int EMB       = 256;
constexpr int D_INNER   = 512;
constexpr int HEADDIM   = 64;
constexpr int NHEADS    = 8;
constexpr int DSTATE    = 128;
constexpr int CONV_DIM  = 768;    // D_INNER + 2*DSTATE
constexpr int D_IN_PROJ = 1288;   // 2*D_INNER + 2*DSTATE + NHEADS
constexpr int SEQ       = 64;
constexpr int LSEQ      = 512;
constexpr int NTOK      = SEQ * LSEQ;            // 32768
constexpr int NPAD_INPROJ = ((D_IN_PROJ + 127) / 128) * 128;  // 1408

// ---------------------------------------------------------------------------
// WMMA fragment loaders (layouts per CDNA5 ISA §7.12.2)
// A 16x32 f16: lane<16 holds K {0..7,16..23}; lane>=16 holds K {8..15,24..31}
// B 32x16 f16 (col n per lane): lane<16 holds K {0..7,16..23}; lane>=16 holds K {8..15,24..31}
// ---------------------------------------------------------------------------
__device__ __forceinline__ v16h load_fragA(const _Float16* __restrict__ base,
                                           int ld, int row, int kk, int lane) {
  const _Float16* p = base + (size_t)row * ld + kk + ((lane >> 4) << 3);
  v8h lo = *(const v8h*)(p);
  v8h hi = *(const v8h*)(p + 16);
  return __builtin_shufflevector(lo, hi, 0,1,2,3,4,5,6,7,8,9,10,11,12,13,14,15);
}

__device__ __forceinline__ v16h load_fragB(const _Float16* __restrict__ base,
                                           int ld, int row, int kk, int lane) {
  const _Float16* p = base + (size_t)row * ld + kk + ((lane >> 4) << 3);
  v8h lo = *(const v8h*)(p);
  v8h hi = *(const v8h*)(p + 16);
  return __builtin_shufflevector(lo, hi, 0,1,2,3,4,5,6,7,8,9,10,11,12,13,14,15);
}

// ---------------------------------------------------------------------------
// GEMM: C[M,N](f32) = A[M,K](f16) * W[Npad,K](f16)^T (+bias)(+gelu)
// W is zero-padded to Npad (multiple of 128) rows, so B loads need no guard.
// Block = 256 threads = 8 waves (4M x 2N); wave computes a 16x64 strip with
// one shared A fragment per K-step feeding 4 WMMAs. Block tile = 64M x 128N.
// EPI: 0 = none, 1 = +bias, 2 = +bias + exact GELU
// ---------------------------------------------------------------------------
template<int EPI>
__global__ void gemm_wmma_kernel(const _Float16* __restrict__ A,
                                 const _Float16* __restrict__ W,
                                 const float* __restrict__ bias,
                                 float* __restrict__ C,
                                 int M, int N, int K)
{
  const int lane = threadIdx.x & 31;
  const int wave = threadIdx.x >> 5;                 // 0..7
  const int tile_m  = (blockIdx.y * 4 + (wave >> 1)) * 16;
  const int tile_n0 = (blockIdx.x * 2 + (wave & 1)) * 64;

  const int arow  = tile_m + (lane & 15);
  const int brow0 = tile_n0 + (lane & 15);

  v8f acc0 = {}, acc1 = {}, acc2 = {}, acc3 = {};
  for (int kk = 0; kk < K; kk += 32) {
    const v16h a  = load_fragA(A, K, arow, kk, lane);
    const v16h b0 = load_fragB(W, K, brow0,      kk, lane);
    const v16h b1 = load_fragB(W, K, brow0 + 16, kk, lane);
    const v16h b2 = load_fragB(W, K, brow0 + 32, kk, lane);
    const v16h b3 = load_fragB(W, K, brow0 + 48, kk, lane);
    acc0 = __builtin_amdgcn_wmma_f32_16x16x32_f16(false, a, false, b0, (short)0, acc0, false, false);
    acc1 = __builtin_amdgcn_wmma_f32_16x16x32_f16(false, a, false, b1, (short)0, acc1, false, false);
    acc2 = __builtin_amdgcn_wmma_f32_16x16x32_f16(false, a, false, b2, (short)0, acc2, false, false);
    acc3 = __builtin_amdgcn_wmma_f32_16x16x32_f16(false, a, false, b3, (short)0, acc3, false, false);
  }

  const int m0 = tile_m + ((lane >> 4) << 3);
  const int nl = lane & 15;
  v8f accs[4] = {acc0, acc1, acc2, acc3};
#pragma unroll
  for (int j = 0; j < 4; ++j) {
    const int n = tile_n0 + j * 16 + nl;
    if (n < N) {
      const float bv = (EPI >= 1) ? bias[n] : 0.0f;
#pragma unroll
      for (int r = 0; r < 8; ++r) {
        float v = accs[j][r] + bv;
        if (EPI == 2) v = 0.5f * v * (1.0f + erff(v * 0.70710678118654752f));
        C[(size_t)(m0 + r) * N + n] = v;
      }
    }
  }
}

// ---------------------------------------------------------------------------
// LayerNorm over 256 cols -> f16 out. 8 waves/block, one token per wave.
// ---------------------------------------------------------------------------
__global__ void layernorm_f16_kernel(const float* __restrict__ x,
                                     const float* __restrict__ w,
                                     const float* __restrict__ b,
                                     _Float16* __restrict__ out, int M)
{
  const int wave = threadIdx.x >> 5, lane = threadIdx.x & 31;
  const int row = blockIdx.x * 8 + wave;
  if (row >= M) return;
  const float* xr = x + (size_t)row * EMB;
  float v[8], s = 0.0f;
#pragma unroll
  for (int i = 0; i < 8; ++i) { v[i] = xr[lane + 32 * i]; s += v[i]; }
#pragma unroll
  for (int m = 16; m >= 1; m >>= 1) s += __shfl_xor(s, m, 32);
  const float mean = s * (1.0f / EMB);
  float s2 = 0.0f;
#pragma unroll
  for (int i = 0; i < 8; ++i) { const float d = v[i] - mean; s2 += d * d; }
#pragma unroll
  for (int m = 16; m >= 1; m >>= 1) s2 += __shfl_xor(s2, m, 32);
  const float inv = rsqrtf(s2 * (1.0f / EMB) + 1e-5f);
#pragma unroll
  for (int i = 0; i < 8; ++i) {
    const int c = lane + 32 * i;
    out[(size_t)row * EMB + c] = (_Float16)((v[i] - mean) * inv * w[c] + b[c]);
  }
}

// ---------------------------------------------------------------------------
// Weight prep: f32 -> f16 with zero-padding past `rows`
// ---------------------------------------------------------------------------
__global__ void cast_pad_f16_kernel(const float* __restrict__ in,
                                    _Float16* __restrict__ out,
                                    int rows, int cols, int rows_pad)
{
  const int i = blockIdx.x * 256 + threadIdx.x;
  if (i >= rows_pad * cols) return;
  const int r = i / cols;
  out[i] = (r < rows) ? (_Float16)in[i] : (_Float16)0.0f;
}

__global__ void wn_cast_kernel(const float* __restrict__ g,
                               const float* __restrict__ v,
                               _Float16* __restrict__ out, int rows, int cols)
{
  const int wave = threadIdx.x >> 5, lane = threadIdx.x & 31;
  const int r = blockIdx.x * 8 + wave;
  if (r >= rows) return;
  const float* vr = v + (size_t)r * cols;
  float s = 0.0f;
  for (int c = lane; c < cols; c += 32) { const float x = vr[c]; s += x * x; }
#pragma unroll
  for (int m = 16; m >= 1; m >>= 1) s += __shfl_xor(s, m, 32);
  const float scale = g[r] * rsqrtf(s);
  for (int c = lane; c < cols; c += 32)
    out[(size_t)r * cols + c] = (_Float16)(vr[c] * scale);
}

// ---------------------------------------------------------------------------
// Depthwise causal conv (DCONV=4) + SiLU over xBC slice of zxbcdt
// ---------------------------------------------------------------------------
__global__ void conv_silu_kernel(const float* __restrict__ zx,
                                 const float* __restrict__ cw,
                                 const float* __restrict__ cb,
                                 float* __restrict__ xc)
{
  const int idx = blockIdx.x * 256 + threadIdx.x;
  if (idx >= NTOK * CONV_DIM) return;
  const int c  = idx % CONV_DIM;
  const int tk = idx / CONV_DIM;
  const int t  = tk & (LSEQ - 1);
  float acc = cb[c];
#pragma unroll
  for (int k = 0; k < 4; ++k) {
    const int tt = t - 3 + k;
    if (tt >= 0)
      acc += cw[c * 4 + k] * zx[(size_t)(tk - 3 + k) * D_IN_PROJ + D_INNER + c];
  }
  const float sig = 1.0f / (1.0f + expf(-acc));
  xc[(size_t)tk * CONV_DIM + c] = acc * sig;
}

// softplus(dt + dt_bias)
__global__ void dt_kernel(const float* __restrict__ zx,
                          const float* __restrict__ dt_bias,
                          float* __restrict__ dtv)
{
  const int idx = blockIdx.x * 256 + threadIdx.x;
  if (idx >= NTOK * NHEADS) return;
  const int h = idx & 7;
  const int tk = idx >> 3;
  const float x = zx[(size_t)tk * D_IN_PROJ + (D_INNER + CONV_DIM) + h] + dt_bias[h];
  dtv[idx] = (x > 20.0f) ? x : log1pf(expf(x));
}

// ---------------------------------------------------------------------------
// Selective-state scan. One block per (seq, head); 64x128 state in registers
// (32 f32 per thread), B/C/x staged in LDS each step, 512 sequential steps.
// ---------------------------------------------------------------------------
__global__ void ssm_scan_kernel(const float* __restrict__ xc,
                                const float* __restrict__ dtv,
                                const float* __restrict__ A_log,
                                const float* __restrict__ D_param,
                                float* __restrict__ y)
{
  const int b  = blockIdx.x >> 3;
  const int hd = blockIdx.x & 7;
  const int tid = threadIdx.x;
  const int p = tid & 63;
  const int n0 = (tid >> 6) * 32;
  __shared__ float sB[DSTATE], sC[DSTATE], sx[HEADDIM], sy[256];

  float h[32];
#pragma unroll
  for (int i = 0; i < 32; ++i) h[i] = 0.0f;

  const float Ah = -expf(A_log[hd]);
  const float Dh = D_param[hd];
  const size_t base = (size_t)b * LSEQ;

  for (int t = 0; t < LSEQ; ++t) {
    const size_t tk = base + t;
    const float* row = xc + tk * CONV_DIM;
    if (tid < 128) sB[tid] = row[D_INNER + tid];
    else           sC[tid - 128] = row[D_INNER + DSTATE + (tid - 128)];
    if (tid < 64)  sx[tid] = row[hd * HEADDIM + tid];
    const float dt = dtv[tk * NHEADS + hd];
    __syncthreads();

    const float dA  = expf(dt * Ah);
    const float dtx = dt * sx[p];
    float acc = 0.0f;
#pragma unroll
    for (int i = 0; i < 32; ++i) {
      h[i] = h[i] * dA + dtx * sB[n0 + i];
      acc += h[i] * sC[n0 + i];
    }
    sy[tid] = acc;
    __syncthreads();

    if (tid < 64) {
      const float yv = sy[tid] + sy[tid + 64] + sy[tid + 128] + sy[tid + 192]
                     + Dh * sx[tid];
      y[tk * D_INNER + hd * HEADDIM + tid] = yv;
    }
    __syncthreads();
  }
}

// ---------------------------------------------------------------------------
// yg = y * silu(z_gate); RMS-norm over 512; f16 out. One block per token.
// ---------------------------------------------------------------------------
__global__ void gate_rms_kernel(const float* __restrict__ y,
                                const float* __restrict__ zx,
                                const float* __restrict__ nw,
                                _Float16* __restrict__ out)
{
  const int tk = blockIdx.x;
  const int tid = threadIdx.x;
  __shared__ float red[8];
  float vals[2];
  float s2 = 0.0f;
#pragma unroll
  for (int i = 0; i < 2; ++i) {
    const int c = tid + i * 256;
    const float zg = zx[(size_t)tk * D_IN_PROJ + c];
    const float sig = 1.0f / (1.0f + expf(-zg));
    const float yg = y[(size_t)tk * D_INNER + c] * zg * sig;
    vals[i] = yg;
    s2 += yg * yg;
  }
#pragma unroll
  for (int m = 16; m >= 1; m >>= 1) s2 += __shfl_xor(s2, m, 32);
  if ((tid & 31) == 0) red[tid >> 5] = s2;
  __syncthreads();
  float tot = 0.0f;
#pragma unroll
  for (int w = 0; w < 8; ++w) tot += red[w];
  const float inv = rsqrtf(tot * (1.0f / D_INNER) + 1e-5f);
#pragma unroll
  for (int i = 0; i < 2; ++i) {
    const int c = tid + i * 256;
    out[(size_t)tk * D_INNER + c] = (_Float16)(vals[i] * inv * nw[c]);
  }
}

// ---------------------------------------------------------------------------
// Elementwise glue
// ---------------------------------------------------------------------------
__global__ void cast32to16_kernel(const float* __restrict__ in,
                                  _Float16* __restrict__ out, int n)
{
  const int i = blockIdx.x * 256 + threadIdx.x;
  if (i < n) out[i] = (_Float16)in[i];
}

__global__ void add_kernel(float* __restrict__ dst,
                           const float* __restrict__ a,
                           const float* __restrict__ b, int n)
{
  const int i = blockIdx.x * 256 + threadIdx.x;
  if (i < n) dst[i] = a[i] + b[i];
}

// ---------------------------------------------------------------------------
extern "C" void kernel_launch(void* const* d_in, const int* in_sizes, int n_in,
                              void* d_out, int out_size, void* d_ws, size_t ws_size,
                              hipStream_t stream)
{
  const float* z          = (const float*)d_in[0];
  const float* ln1_w      = (const float*)d_in[1];
  const float* ln1_b      = (const float*)d_in[2];
  const float* in_proj_w  = (const float*)d_in[3];
  const float* conv_w     = (const float*)d_in[4];
  const float* conv_b     = (const float*)d_in[5];
  const float* dt_bias    = (const float*)d_in[6];
  const float* A_log      = (const float*)d_in[7];
  const float* D_param    = (const float*)d_in[8];
  const float* ssm_norm_w = (const float*)d_in[9];
  const float* out_proj_w = (const float*)d_in[10];
  const float* ff_g       = (const float*)d_in[11];
  const float* ff_v       = (const float*)d_in[12];
  const float* ff_b       = (const float*)d_in[13];
  const float* ln2_w      = (const float*)d_in[14];
  const float* ln2_b      = (const float*)d_in[15];
  const float* fc1_g      = (const float*)d_in[16];
  const float* fc1_v      = (const float*)d_in[17];
  const float* fc1_b      = (const float*)d_in[18];
  const float* fc2_g      = (const float*)d_in[19];
  const float* fc2_v      = (const float*)d_in[20];
  const float* fc2_b      = (const float*)d_in[21];
  float* out = (float*)d_out;

  // ---- workspace carve-up (256B aligned) ----
  char* ws = (char*)d_ws;
  size_t off = 0;
  auto carve = [&](size_t bytes) -> char* {
    off = (off + 255) & ~(size_t)255;
    char* p = ws + off;
    off += bytes;
    return p;
  };
  _Float16* w16_inproj  = (_Float16*)carve((size_t)NPAD_INPROJ * EMB * 2);
  _Float16* w16_outproj = (_Float16*)carve((size_t)EMB * D_INNER * 2);
  _Float16* w16_ff      = (_Float16*)carve((size_t)EMB * EMB * 2);
  _Float16* w16_fc1     = (_Float16*)carve((size_t)EMB * EMB * 2);
  _Float16* w16_fc2     = (_Float16*)carve((size_t)EMB * EMB * 2);
  _Float16* u16         = (_Float16*)carve((size_t)NTOK * EMB * 2);
  float*    zxb         = (float*)   carve((size_t)NTOK * D_IN_PROJ * 4);
  char*     xc_region   =            carve((size_t)NTOK * CONV_DIM * 4);
  float*    dtv         = (float*)   carve((size_t)NTOK * NHEADS * 4);
  float*    ybuf        = (float*)   carve((size_t)NTOK * D_INNER * 4);
  _Float16* yn16        = (_Float16*)carve((size_t)NTOK * D_INNER * 2);
  // post-scan temporaries overlay the (then-dead) conv buffer: 84MB <= 100MB
  float*    xconv = (float*)xc_region;
  float*    t0    = (float*)xc_region;                                 // f32 tmp
  float*    t1    = (float*)(xc_region + (size_t)NTOK * EMB * 4);      // f32 tmp
  _Float16* t16   = (_Float16*)(xc_region + (size_t)NTOK * EMB * 8);   // f16 tmp

  const dim3 blk(256);

  // ---- weight prep (f16, zero-padded rows + weight-norm) ----
  cast_pad_f16_kernel<<<dim3((NPAD_INPROJ * EMB + 255) / 256), blk, 0, stream>>>(
      in_proj_w, w16_inproj, D_IN_PROJ, EMB, NPAD_INPROJ);
  cast_pad_f16_kernel<<<dim3((EMB * D_INNER + 255) / 256), blk, 0, stream>>>(
      out_proj_w, w16_outproj, EMB, D_INNER, EMB);
  wn_cast_kernel<<<dim3(EMB / 8), blk, 0, stream>>>(ff_g,  ff_v,  w16_ff,  EMB, EMB);
  wn_cast_kernel<<<dim3(EMB / 8), blk, 0, stream>>>(fc1_g, fc1_v, w16_fc1, EMB, EMB);
  wn_cast_kernel<<<dim3(EMB / 8), blk, 0, stream>>>(fc2_g, fc2_v, w16_fc2, EMB, EMB);

  // ---- LN1 -> u (f16) ----
  layernorm_f16_kernel<<<dim3(NTOK / 8), blk, 0, stream>>>(z, ln1_w, ln1_b, u16, NTOK);

  // ---- in_proj GEMM: (32768 x 256) x (1288 x 256)^T -> zxbcdt ----
  gemm_wmma_kernel<0><<<dim3(NPAD_INPROJ / 128, NTOK / 64), blk, 0, stream>>>(
      u16, w16_inproj, nullptr, zxb, NTOK, D_IN_PROJ, EMB);

  // ---- conv + SiLU, dt softplus ----
  conv_silu_kernel<<<dim3((NTOK * CONV_DIM + 255) / 256), blk, 0, stream>>>(
      zxb, conv_w, conv_b, xconv);
  dt_kernel<<<dim3((NTOK * NHEADS + 255) / 256), blk, 0, stream>>>(zxb, dt_bias, dtv);

  // ---- SSM scan: 512 blocks (seq x head) ----
  ssm_scan_kernel<<<dim3(SEQ * NHEADS), blk, 0, stream>>>(xconv, dtv, A_log,
                                                          D_param, ybuf);

  // ---- gate + RMS norm -> yn (f16) ----
  gate_rms_kernel<<<dim3(NTOK), blk, 0, stream>>>(ybuf, zxb, ssm_norm_w, yn16);

  // ---- out_proj GEMM: (32768 x 512) x (256 x 512)^T -> t0 ----
  gemm_wmma_kernel<0><<<dim3(EMB / 128, NTOK / 64), blk, 0, stream>>>(
      yn16, w16_outproj, nullptr, t0, NTOK, EMB, D_INNER);

  // ---- ff (weight-norm) GEMM + bias, then z-residual ----
  cast32to16_kernel<<<dim3(NTOK * EMB / 256), blk, 0, stream>>>(t0, t16, NTOK * EMB);
  gemm_wmma_kernel<1><<<dim3(EMB / 128, NTOK / 64), blk, 0, stream>>>(
      t16, w16_ff, ff_b, t1, NTOK, EMB, EMB);
  add_kernel<<<dim3(NTOK * EMB / 256), blk, 0, stream>>>(t1, t1, z, NTOK * EMB);

  // ---- LN2 -> h (f16); fc1 + GELU; fc2 + bias; final residual ----
  layernorm_f16_kernel<<<dim3(NTOK / 8), blk, 0, stream>>>(t1, ln2_w, ln2_b, t16, NTOK);
  gemm_wmma_kernel<2><<<dim3(EMB / 128, NTOK / 64), blk, 0, stream>>>(
      t16, w16_fc1, fc1_b, t0, NTOK, EMB, EMB);
  cast32to16_kernel<<<dim3(NTOK * EMB / 256), blk, 0, stream>>>(t0, t16, NTOK * EMB);
  gemm_wmma_kernel<1><<<dim3(EMB / 128, NTOK / 64), blk, 0, stream>>>(
      t16, w16_fc2, fc2_b, t0, NTOK, EMB, EMB);
  add_kernel<<<dim3(NTOK * EMB / 256), blk, 0, stream>>>(out, t1, t0, NTOK * EMB);

  (void)in_sizes; (void)n_in; (void)out_size; (void)ws_size;
}
